// Qwen2VLVisionModel_16097537425640
// MI455X (gfx1250) — matrix operand order repaired and checked by reading, not compiled, not run on hardware
//
#include <hip/hip_runtime.h>
#include <hip/hip_bf16.h>

// ---------------------------------------------------------------------------
// Qwen2-VL vision tower on gfx1250 (MI455X), bf16 WMMA everywhere.
// S=1024, D=1280, H=16, HD=80 (pad 96), DEPTH=8.
// GEMM staging uses CDNA5 GLOBAL_LOAD_ASYNC_TO_LDS_B128 (ASYNCcnt) with LDS
// double buffering (last K-tile peeled so the steady-state loop is
// branch-free); fragments read back via ds_load_b128; math via
// v_wmma_f32_16x16x32_bf16.
// ---------------------------------------------------------------------------

typedef __attribute__((ext_vector_type(16))) __bf16 bf16x16;
typedef __attribute__((ext_vector_type(8)))  float  f32x8;

#define TILE_M 128
#define TILE_N 128
#define TILE_K 32
#define LDS_LD (TILE_K + 8)   // bf16 elements; row pitch 80B (16B-aligned, bank-skewed)

static __device__ __forceinline__ unsigned short f32_to_bf16(float f) {
    unsigned int u = __float_as_uint(f);
    u += 0x7FFFu + ((u >> 16) & 1u);   // round-to-nearest-even
    return (unsigned short)(u >> 16);
}

// LDS byte offset of a generic pointer to __shared__ (low 32 bits of the
// aperture-based generic address are the LDS offset).
static __device__ __forceinline__ unsigned lds_off32(const void* p) {
    return (unsigned)(unsigned long long)p;
}

// One 64B row chunk: 4x async b128, INST_OFFSET advances LDS and global
// addresses in lockstep (dsaddr = VDST + off, maddr = VADDR + off).
static __device__ __forceinline__ void async_row64B(unsigned lds,
                                                    const unsigned short* g) {
    asm volatile(
        "global_load_async_to_lds_b128 %0, %1, off\n\t"
        "global_load_async_to_lds_b128 %0, %1, off offset:16\n\t"
        "global_load_async_to_lds_b128 %0, %1, off offset:32\n\t"
        "global_load_async_to_lds_b128 %0, %1, off offset:48"
        :: "v"(lds), "v"(g) : "memory");
}

// ---------------------------------------------------------------------------
// Generic fused GEMM:  C[b] = act(alpha * A[b] @ W[b]^T + bias + Res[b])
// A: [M,K] bf16 (row major, lda), W: [N,K] bf16 (row major, ldw)
// ACT: 0=none, 1=tanh-gelu, 2=erf-gelu. K must be a multiple of 32.
// OOB rows are clamped (not zeroed): they only feed OOB outputs, never stored.
// ---------------------------------------------------------------------------
template <int ACT, bool HAS_BIAS, bool HAS_RES, bool WRITE_F32, bool WRITE_BF16>
__global__ __launch_bounds__(256) void gemm_bf16_wmma(
    const unsigned short* __restrict__ A, long long sA, int lda,
    const unsigned short* __restrict__ W, long long sW, int ldw,
    const float* __restrict__ bias,
    float* __restrict__ C, long long sC, int ldc,
    unsigned short* __restrict__ Cb, long long sCb, int ldcb,
    const float* __restrict__ Res, long long sR, int ldr,
    int M, int N, int K, float alpha)
{
    __shared__ unsigned short As[2][TILE_M * LDS_LD];
    __shared__ unsigned short Ws[2][TILE_N * LDS_LD];

    const int tid  = threadIdx.x;
    const int lane = tid & 31;
    const int wave = tid >> 5;
    const int wm   = (wave & 3) * 32;   // 4 waves along M
    const int wn   = (wave >> 2) * 64;  // 2 waves along N
    const int bm   = blockIdx.y * TILE_M;
    const int bn   = blockIdx.x * TILE_N;
    const int batch = blockIdx.z;

    const unsigned short* Ab = A + (long long)batch * sA;
    const unsigned short* Wb = W + (long long)batch * sW;

    // --- per-thread async staging assignment: one 32-elem row per tile ---
    const int  r   = tid & 127;
    const bool isA = tid < 128;
    int grow = isA ? (bm + r) : (bn + r);
    {   // clamp OOB rows to a valid row (branch-free; EXEC stays all-ones)
        const int lim = (isA ? M : N) - 1;
        grow = grow < lim ? grow : lim;
    }
    const unsigned short* gptr =
        isA ? (Ab + (long long)grow * lda) : (Wb + (long long)grow * ldw);
    const unsigned ldsRow =
        (isA ? lds_off32(&As[0][0]) : lds_off32(&Ws[0][0])) +
        (unsigned)r * (LDS_LD * 2);
    const unsigned bufStride = (unsigned)(TILE_M * LDS_LD * 2);  // bytes

    f32x8 acc[2][4];
    const f32x8 zero = {0.f, 0.f, 0.f, 0.f, 0.f, 0.f, 0.f, 0.f};
#pragma unroll
    for (int i = 0; i < 2; ++i)
#pragma unroll
        for (int j = 0; j < 4; ++j) acc[i][j] = zero;

    const int row16 = lane & 15;
    const int ahalf = (lane >> 4) * 8;   // A frag: K {0..7,16..23} / {8..15,24..31}
    const int bhalf = (lane >> 4) * 8;   // B frag: K {0..7,16..23} / {8..15,24..31}

    union FragU { uint4 q[2]; bf16x16 v; };

    auto compute_tile = [&](int cur) {
        const unsigned short* Ac = &As[cur][0];
        const unsigned short* Wc = &Ws[cur][0];
        FragU afr[2], wfr[4];
#pragma unroll
        for (int i = 0; i < 2; ++i) {
            const int rr = wm + i * 16 + row16;
            afr[i].q[0] = *(const uint4*)&Ac[rr * LDS_LD + ahalf];
            afr[i].q[1] = *(const uint4*)&Ac[rr * LDS_LD + ahalf + 16];
        }
#pragma unroll
        for (int j = 0; j < 4; ++j) {
            const int rr = wn + j * 16 + row16;
            wfr[j].q[0] = *(const uint4*)&Wc[rr * LDS_LD + bhalf];
            wfr[j].q[1] = *(const uint4*)&Wc[rr * LDS_LD + bhalf + 16];
        }
#pragma unroll
        for (int i = 0; i < 2; ++i)
#pragma unroll
            for (int j = 0; j < 4; ++j)
                acc[i][j] = __builtin_amdgcn_wmma_f32_16x16x32_bf16(
                    false, afr[i].v, false, wfr[j].v, (short)0, acc[i][j],
                    false, false);
    };

    const int ntiles = K / TILE_K;
    async_row64B(ldsRow, gptr);          // prologue: tile 0 -> buffer 0

    // steady state: unconditionally prefetch tile t+1, compute tile t
    int t = 0;
    for (; t < ntiles - 1; ++t) {
        const int cur = t & 1;
        async_row64B(ldsRow + (cur ^ 1) * bufStride,
                     gptr + (long long)(t + 1) * TILE_K);
        asm volatile("s_wait_asynccnt 0x4" ::: "memory");  // tile t landed
        __syncthreads();           // all waves' tile-t data visible
        compute_tile(cur);
        __syncthreads();           // all waves done with buf cur
    }
    // peeled last tile: nothing left to prefetch, no trailing barrier
    asm volatile("s_wait_asynccnt 0x0" ::: "memory");
    __syncthreads();
    compute_tile(t & 1);

    // --- epilogue: C/D layout -> lane 0-15: M=r,N=lane ; lane 16-31: M=8+r ---
    const int nj    = lane & 15;
    const int mbase = (lane < 16) ? 0 : 8;
#pragma unroll
    for (int i = 0; i < 2; ++i) {
#pragma unroll
        for (int j = 0; j < 4; ++j) {
            const int n = bn + wn + j * 16 + nj;
            if (n >= N) continue;
            const float bv = HAS_BIAS ? bias[n] : 0.f;
#pragma unroll
            for (int rr = 0; rr < 8; ++rr) {
                const int m = bm + wm + i * 16 + mbase + rr;
                if (m >= M) continue;
                float v = alpha * acc[i][j][rr] + bv;
                if (HAS_RES)
                    v += Res[(long long)batch * sR + (long long)m * ldr + n];
                if (ACT == 1) {
                    const float x = v;
                    v = 0.5f * x *
                        (1.f + tanhf(0.7978845608028654f *
                                     (x + 0.044715f * x * x * x)));
                } else if (ACT == 2) {
                    const float x = v;
                    v = 0.5f * x * (1.f + erff(x * 0.7071067811865476f));
                }
                if (WRITE_F32)
                    C[(long long)batch * sC + (long long)m * ldc + n] = v;
                if (WRITE_BF16)
                    Cb[(long long)batch * sCb + (long long)m * ldcb + n] =
                        f32_to_bf16(v);
            }
        }
    }
}

// ---------------------------------------------------------------------------
// fp32 -> bf16 conversion with optional column padding (zeros)
// ---------------------------------------------------------------------------
__global__ __launch_bounds__(256) void cvt_pad_bf16(
    const float* __restrict__ src, unsigned short* __restrict__ dst,
    long long rows, int cols, int colsp)
{
    const long long i = (long long)blockIdx.x * blockDim.x + threadIdx.x;
    const long long total = rows * (long long)colsp;
    if (i >= total) return;
    const long long r = i / colsp;
    const int c = (int)(i - r * colsp);
    dst[i] = (c < cols) ? f32_to_bf16(src[r * (long long)cols + c])
                        : (unsigned short)0;
}

// ---------------------------------------------------------------------------
// LayerNorm (eps=1e-6) over D=1280, bf16 output. One block per row.
// ---------------------------------------------------------------------------
__global__ __launch_bounds__(256) void layernorm_bf16(
    const float* __restrict__ x, const float* __restrict__ w,
    const float* __restrict__ b, unsigned short* __restrict__ out, int D_)
{
    const int row = blockIdx.x;
    const float* xr = x + (long long)row * D_;
    __shared__ float red[256];
    float s = 0.f;
    for (int i = threadIdx.x; i < D_; i += 256) s += xr[i];
    red[threadIdx.x] = s;
    __syncthreads();
    for (int off = 128; off > 0; off >>= 1) {
        if (threadIdx.x < off) red[threadIdx.x] += red[threadIdx.x + off];
        __syncthreads();
    }
    const float mean = red[0] / D_;
    __syncthreads();
    float v = 0.f;
    for (int i = threadIdx.x; i < D_; i += 256) {
        const float d = xr[i] - mean;
        v += d * d;
    }
    red[threadIdx.x] = v;
    __syncthreads();
    for (int off = 128; off > 0; off >>= 1) {
        if (threadIdx.x < off) red[threadIdx.x] += red[threadIdx.x + off];
        __syncthreads();
    }
    const float rstd = rsqrtf(red[0] / D_ + 1e-6f);
    for (int i = threadIdx.x; i < D_; i += 256) {
        const float y = (xr[i] - mean) * rstd * w[i] + b[i];
        out[(long long)row * D_ + i] = f32_to_bf16(y);
    }
}

// ---------------------------------------------------------------------------
// RoPE cos/sin tables for 32x32 grid, merge=2 (1024 positions x 80 dims)
// ---------------------------------------------------------------------------
__global__ void rope_cossin(float* __restrict__ cosb, float* __restrict__ sinb)
{
    const int p = blockIdx.x * blockDim.x + threadIdx.x;
    if (p >= 1024) return;
    const int dj = p & 1, di = (p >> 1) & 1, j = (p >> 2) & 15,
              i = (p >> 6) & 15;
    const int hp = i * 2 + di, wp = j * 2 + dj;
    for (int t = 0; t < 20; ++t) {
        const float invf = powf(10000.f, -(float)(2 * t) / 40.f);
        const float fh = hp * invf, fw = wp * invf;
        const float ch = cosf(fh), sh = sinf(fh);
        const float cw = cosf(fw), sw = sinf(fw);
        cosb[p * 80 + t]      = ch; sinb[p * 80 + t]      = sh;
        cosb[p * 80 + 20 + t] = cw; sinb[p * 80 + 20 + t] = sw;
        cosb[p * 80 + 40 + t] = ch; sinb[p * 80 + 40 + t] = sh;
        cosb[p * 80 + 60 + t] = cw; sinb[p * 80 + 60 + t] = sw;
    }
}

// ---------------------------------------------------------------------------
// Apply RoPE to q,k; pack q,k -> bf16 [H][S][96] (pad 80->96), v -> bf16
// transposed [H][80][S]. One block per token s.
// ---------------------------------------------------------------------------
__global__ __launch_bounds__(256) void rope_pack(
    const float* __restrict__ qkv, const float* __restrict__ cosb,
    const float* __restrict__ sinb, unsigned short* __restrict__ qo,
    unsigned short* __restrict__ ko, unsigned short* __restrict__ vt)
{
    const int s = blockIdx.x;
    const float* base = qkv + (long long)s * 3840;
    for (int idx = threadIdx.x; idx < 16 * 96; idx += 256) {
        const int h = idx / 96, d = idx - h * 96;
        const long long qk_off = ((long long)h * 1024 + s) * 96 + d;
        if (d >= 80) { qo[qk_off] = 0; ko[qk_off] = 0; continue; }
        const float c  = cosb[s * 80 + d];
        const float sn = sinb[s * 80 + d];
        const float q  = base[h * 80 + d];
        const float k  = base[1280 + h * 80 + d];
        const float v  = base[2560 + h * 80 + d];
        const float qr = (d < 40) ? -base[h * 80 + d + 40]
                                  :  base[h * 80 + d - 40];
        const float kr = (d < 40) ? -base[1280 + h * 80 + d + 40]
                                  :  base[1280 + h * 80 + d - 40];
        qo[qk_off] = f32_to_bf16(q * c + qr * sn);
        ko[qk_off] = f32_to_bf16(k * c + kr * sn);
        vt[((long long)h * 80 + d) * 1024 + s] = f32_to_bf16(v);
    }
}

// ---------------------------------------------------------------------------
// Row softmax (T=1024) fp32 -> bf16 probs. One block per (head,row).
// ---------------------------------------------------------------------------
__global__ __launch_bounds__(256) void softmax_bf16(
    const float* __restrict__ scores, unsigned short* __restrict__ probs,
    int T)
{
    const long long row = blockIdx.x;
    const float* sr = scores + row * T;
    __shared__ float red[256];
    float mx = -3.4e38f;
    for (int i = threadIdx.x; i < T; i += 256) mx = fmaxf(mx, sr[i]);
    red[threadIdx.x] = mx;
    __syncthreads();
    for (int off = 128; off > 0; off >>= 1) {
        if (threadIdx.x < off)
            red[threadIdx.x] = fmaxf(red[threadIdx.x], red[threadIdx.x + off]);
        __syncthreads();
    }
    mx = red[0];
    __syncthreads();
    float s = 0.f;
    for (int i = threadIdx.x; i < T; i += 256) s += __expf(sr[i] - mx);
    red[threadIdx.x] = s;
    __syncthreads();
    for (int off = 128; off > 0; off >>= 1) {
        if (threadIdx.x < off) red[threadIdx.x] += red[threadIdx.x + off];
        __syncthreads();
    }
    const float inv = 1.f / red[0];
    for (int i = threadIdx.x; i < T; i += 256)
        probs[row * T + i] = f32_to_bf16(__expf(sr[i] - mx) * inv);
}

// ---------------------------------------------------------------------------
// Host orchestration
// ---------------------------------------------------------------------------
extern "C" void kernel_launch(void* const* d_in, const int* in_sizes, int n_in,
                              void* d_out, int out_size, void* d_ws,
                              size_t ws_size, hipStream_t stream)
{
    (void)in_sizes; (void)n_in; (void)out_size; (void)ws_size;
    const float* pix     = (const float*)d_in[0];
    const float* w_patch = (const float*)d_in[1];
    const float* qkv_w   = (const float*)d_in[2];
    const float* qkv_b   = (const float*)d_in[3];
    const float* proj_w  = (const float*)d_in[4];
    const float* proj_b  = (const float*)d_in[5];
    const float* fc1_w   = (const float*)d_in[6];
    const float* fc1_b   = (const float*)d_in[7];
    const float* fc2_w   = (const float*)d_in[8];
    const float* fc2_b   = (const float*)d_in[9];
    const float* ln1_w   = (const float*)d_in[10];
    const float* ln1_b   = (const float*)d_in[11];
    const float* ln2_w   = (const float*)d_in[12];
    const float* ln2_b   = (const float*)d_in[13];
    const float* mnorm_w = (const float*)d_in[14];
    const float* mnorm_b = (const float*)d_in[15];
    const float* mfc1_w  = (const float*)d_in[16];
    const float* mfc1_b  = (const float*)d_in[17];
    const float* mfc2_w  = (const float*)d_in[18];
    const float* mfc2_b  = (const float*)d_in[19];
    float* out = (float*)d_out;

    const int S = 1024, D = 1280, H3 = 3840, D4 = 5120, KP = 1184;

    char* wsp = (char*)d_ws;
    auto alloc = [&](size_t bytes) -> char* {
        char* p = wsp;
        wsp += (bytes + 255) & ~(size_t)255;
        return p;
    };
    unsigned short* wpatch_bf = (unsigned short*)alloc((size_t)D * KP * 2);
    unsigned short* qkvw_bf   = (unsigned short*)alloc((size_t)8 * H3 * D * 2);
    unsigned short* projw_bf  = (unsigned short*)alloc((size_t)8 * D * D * 2);
    unsigned short* fc1w_bf   = (unsigned short*)alloc((size_t)8 * D4 * D * 2);
    unsigned short* fc2w_bf   = (unsigned short*)alloc((size_t)8 * D * D4 * 2);
    unsigned short* mfc1_bf   = (unsigned short*)alloc((size_t)D4 * D4 * 2);
    unsigned short* mfc2_bf   = (unsigned short*)alloc((size_t)1536 * D4 * 2);
    unsigned short* pix_bf    = (unsigned short*)alloc((size_t)S * KP * 2);
    float*          x         = (float*)alloc((size_t)S * D * 4);
    unsigned short* hbf       = (unsigned short*)alloc((size_t)S * D * 2);
    float*          qkvf      = (float*)alloc((size_t)S * H3 * 4);
    unsigned short* qbf       = (unsigned short*)alloc((size_t)16 * S * 96 * 2);
    unsigned short* kbf       = (unsigned short*)alloc((size_t)16 * S * 96 * 2);
    unsigned short* vt        = (unsigned short*)alloc((size_t)16 * 80 * S * 2);
    float*          scores    = (float*)alloc((size_t)16 * S * S * 4);
    unsigned short* probs     = (unsigned short*)alloc((size_t)16 * S * S * 2);
    float*          o         = (float*)alloc((size_t)S * D * 4);
    unsigned short* obf       = (unsigned short*)alloc((size_t)S * D * 2);
    unsigned short* h4bf      = (unsigned short*)alloc((size_t)S * D4 * 2);
    unsigned short* m1bf      = (unsigned short*)alloc((size_t)256 * D4 * 2);
    float*          cosb      = (float*)alloc((size_t)S * 80 * 4);
    float*          sinb      = (float*)alloc((size_t)S * 80 * 4);

    auto cvt = [&](const float* src, unsigned short* dst, long long rows,
                   int cols, int colsp) {
        const long long total = rows * (long long)colsp;
        const int blocks = (int)((total + 255) / 256);
        cvt_pad_bf16<<<blocks, 256, 0, stream>>>(src, dst, rows, cols, colsp);
    };

    // --- per-launch weight conversion to bf16 ---
    cvt(w_patch, wpatch_bf, D, 1176, KP);
    cvt(pix,     pix_bf,    S, 1176, KP);
    cvt(qkv_w,   qkvw_bf,  (long long)8 * H3, D, D);
    cvt(proj_w,  projw_bf, (long long)8 * D, D, D);
    cvt(fc1_w,   fc1w_bf,  (long long)8 * D4, D, D);
    cvt(fc2_w,   fc2w_bf,  (long long)8 * D, D4, D4);
    cvt(mfc1_w,  mfc1_bf,  D4, D4, D4);
    cvt(mfc2_w,  mfc2_bf,  1536, D4, D4);
    rope_cossin<<<4, 256, 0, stream>>>(cosb, sinb);

    auto grid_for = [](int M, int N, int B) {
        return dim3((unsigned)((N + TILE_N - 1) / TILE_N),
                    (unsigned)((M + TILE_M - 1) / TILE_M), (unsigned)B);
    };

    // --- patch embed: x = pix @ w_patch^T  (M=1024,N=1280,K=1184) ---
    gemm_bf16_wmma<0, false, false, true, false>
        <<<grid_for(S, D, 1), 256, 0, stream>>>(
            pix_bf, 0, KP, wpatch_bf, 0, KP, nullptr,
            x, 0, D, nullptr, 0, 0, nullptr, 0, 0, S, D, KP, 1.0f);

    const float attn_scale = 0.11180339887498948f;  // 80^-0.5

    for (int l = 0; l < 8; ++l) {
        // LN1 -> bf16 h
        layernorm_bf16<<<S, 256, 0, stream>>>(x, ln1_w + l * D, ln1_b + l * D,
                                              hbf, D);
        // qkv = h @ qkv_w^T + qkv_b   (M=1024,N=3840,K=1280)
        gemm_bf16_wmma<0, true, false, true, false>
            <<<grid_for(S, H3, 1), 256, 0, stream>>>(
                hbf, 0, D, qkvw_bf + (long long)l * H3 * D, 0, D,
                qkv_b + (long long)l * H3, qkvf, 0, H3, nullptr, 0, 0,
                nullptr, 0, 0, S, H3, D, 1.0f);
        // RoPE + pack per-head q,k (pad 96) and v transposed
        rope_pack<<<S, 256, 0, stream>>>(qkvf, cosb, sinb, qbf, kbf, vt);
        // scores[h] = scale * q[h] @ k[h]^T  (batch=16, M=N=1024, K=96)
        gemm_bf16_wmma<0, false, false, true, false>
            <<<grid_for(S, S, 16), 256, 0, stream>>>(
                qbf, (long long)S * 96, 96, kbf, (long long)S * 96, 96,
                nullptr, scores, (long long)S * S, S, nullptr, 0, 0,
                nullptr, 0, 0, S, S, 96, attn_scale);
        // softmax -> bf16 probs
        softmax_bf16<<<16 * S, 256, 0, stream>>>(scores, probs, S);
        // o[h] = probs[h] @ vt[h]^T  (M=1024,N=80,K=1024), interleave into [S,1280]
        gemm_bf16_wmma<0, false, false, true, true>
            <<<grid_for(S, 80, 16), 256, 0, stream>>>(
                probs, (long long)S * S, S, vt, (long long)80 * S, S, nullptr,
                o, 80, D, obf, 80, D, nullptr, 0, 0, S, 80, S, 1.0f);
        // x = x + o @ proj_w^T + proj_b
        gemm_bf16_wmma<0, true, true, true, false>
            <<<grid_for(S, D, 1), 256, 0, stream>>>(
                obf, 0, D, projw_bf + (long long)l * D * D, 0, D,
                proj_b + (long long)l * D, x, 0, D, nullptr, 0, 0,
                x, 0, D, S, D, D, 1.0f);
        // LN2 -> bf16 h
        layernorm_bf16<<<S, 256, 0, stream>>>(x, ln2_w + l * D, ln2_b + l * D,
                                              hbf, D);
        // h4 = gelu_tanh(h @ fc1_w^T + fc1_b) -> bf16
        gemm_bf16_wmma<1, true, false, false, true>
            <<<grid_for(S, D4, 1), 256, 0, stream>>>(
                hbf, 0, D, fc1w_bf + (long long)l * D4 * D, 0, D,
                fc1_b + (long long)l * D4, nullptr, 0, 0, h4bf, 0, D4,
                nullptr, 0, 0, S, D4, D, 1.0f);
        // x = x + h4 @ fc2_w^T + fc2_b
        gemm_bf16_wmma<0, true, true, true, false>
            <<<grid_for(S, D, 1), 256, 0, stream>>>(
                h4bf, 0, D4, fc2w_bf + (long long)l * D * D4, 0, D4,
                fc2_b + (long long)l * D, x, 0, D, nullptr, 0, 0,
                x, 0, D, S, D, D4, 1.0f);
    }

    // --- merger: LN -> view [256,5120] -> gelu_erf(mfc1) -> mfc2 -> out ---
    layernorm_bf16<<<S, 256, 0, stream>>>(x, mnorm_w, mnorm_b, hbf, D);
    gemm_bf16_wmma<2, true, false, false, true>
        <<<grid_for(256, D4, 1), 256, 0, stream>>>(
            hbf, 0, D4, mfc1_bf, 0, D4, mfc1_b, nullptr, 0, 0,
            m1bf, 0, D4, nullptr, 0, 0, 256, D4, D4, 1.0f);
    gemm_bf16_wmma<0, true, false, true, false>
        <<<grid_for(256, 1536, 1), 256, 0, stream>>>(
            m1bf, 0, D4, mfc2_bf, 0, D4, mfc2_b, out, 0, 1536,
            nullptr, 0, 0, nullptr, 0, 0, 256, 1536, D4, 1.0f);
}
